// OTLayer_51135880626691
// MI455X (gfx1250) — matrix operand order repaired and checked by reading, not compiled, not run on hardware
//
#include <hip/hip_runtime.h>

// ---------------------------------------------------------------------------
// Sinkhorn OT on MI455X (gfx1250), flash-style fused logsumexp over a
// recomputed cost matrix.
//
//   log_K[i,j] = (2*dot(X_i,Y_j) - |X_i|^2 - |Y_j|^2) / EPS
//   f[i] = log(1/n) - LSE_j(log_K[i,j] + g[j])      (row pass)
//   g[j] = log(1/n) - LSE_i(log_K[i,j] + f[i])      (col pass, X<->Y swap)
//
// The 16x16 cost tile is produced by two v_wmma_f32_16x16x32_f16 (K=64).
// Recomputing log_K keeps traffic to the 2 MiB of f16 X/Y (L2-resident)
// instead of streaming a 256 MiB matrix 200x from HBM (23.3 TB/s -> ~2.2 ms
// just for re-reads; recompute is ~1.7 TFLOP of f16 WMMA, i.e. negligible).
// The tile loop is MANUALLY unrolled x2 (clang ignored the pragma): tile B's
// two WMMAs are issued between tile A's WMMAs and tile A's VALU epilogue, so
// the 4-slot WMMA->VALU hazard window is filled with useful work instead of
// v_nops, and tile A's epilogue covers tile B's hazard window in turn.
// ---------------------------------------------------------------------------

#define N1 8192
#define N2 8192
#define D  64
#define NT (N2 / 16)          // 512 column tiles per pass
#define SINK_ITERS 100
#define INV_EPS 10.0f         // 1/0.1
#define L2E 1.4426950408889634f
#define LN2 0.6931471805599453f
#define K2C (2.0f * INV_EPS * L2E)     // scale on the WMMA dot product
#define LOG_W (-9.01091335f)           // log(1/8192)

typedef _Float16 v8h  __attribute__((ext_vector_type(8)));
typedef _Float16 v16h __attribute__((ext_vector_type(16)));
typedef float    v8f  __attribute__((ext_vector_type(8)));

static __device__ __forceinline__ float fexp2(float x) { return __builtin_amdgcn_exp2f(x); }
static __device__ __forceinline__ float flog2(float x) { return __builtin_amdgcn_logf(x); }

// A-matrix lane layout (16-bit A 16x32): lane L<16 -> row L, K {0-7,16-23};
// lane L>=16 -> row L-16, K {8-15,24-31}. Two contiguous 16B loads per lane.
static __device__ __forceinline__ v16h load_a(const _Float16* __restrict__ R,
                                              int row, int kbase, int half) {
  const v8h lo = *(const v8h*)(R + row * D + kbase + half * 8);
  const v8h hi = *(const v8h*)(R + row * D + kbase + half * 8 + 16);
  return __builtin_shufflevector(lo, hi, 0,1,2,3,4,5,6,7,8,9,10,11,12,13,14,15);
}

// B-matrix lane layout (16-bit B 32x16): lane L<16 -> column L, K {0-7,16-23};
// lane L>=16 -> column L-16, K {8-15,24-31}. Two contiguous 16B loads per lane.
static __device__ __forceinline__ v16h load_b(const _Float16* __restrict__ C,
                                              int n, int kbase, int half) {
  const v8h lo = *(const v8h*)(C + n * D + kbase + half * 8);
  const v8h hi = *(const v8h*)(C + n * D + kbase + half * 8 + 16);
  return __builtin_shufflevector(lo, hi, 0,1,2,3,4,5,6,7,8,9,10,11,12,13,14,15);
}

// --------------------------------------------------------------------------
// prep: f16 copies of X and Y, squared row norms, zero-init g.
// --------------------------------------------------------------------------
__global__ void __launch_bounds__(256)
prep_kernel(const float* __restrict__ X, const float* __restrict__ Y,
            _Float16* __restrict__ Xh, _Float16* __restrict__ Yh,
            float* __restrict__ x2, float* __restrict__ y2,
            float* __restrict__ g) {
  const int idx = blockIdx.x * 256 + threadIdx.x;     // 0 .. 16383
  const float* src;
  _Float16* dst;
  float* sq;
  int r;
  if (idx < N1) {
    r = idx; src = X + (size_t)r * D; dst = Xh + (size_t)r * D; sq = x2;
    g[r] = 0.0f;
  } else {
    r = idx - N1; src = Y + (size_t)r * D; dst = Yh + (size_t)r * D; sq = y2;
  }
  float s = 0.0f;
  #pragma unroll 8
  for (int k = 0; k < D; ++k) {
    float v = src[k];
    s = fmaf(v, v, s);
    dst[k] = (_Float16)v;
  }
  sq[r] = s;
}

// --------------------------------------------------------------------------
// One Sinkhorn half-step: pot_out[row] = LOG_W - LSE_col(log_K + pot_in[col])
// Row pass:  (Rh,Ch,r2,c2) = (Xh,Yh,x2,y2), pot_in=g, pot_out=f
// Col pass:  (Rh,Ch,r2,c2) = (Yh,Xh,y2,x2), pot_in=f, pot_out=g
// Block = 4 waves; each wave owns the same 16-row A tile and strides over
// column tiles (2 per iteration); online (max,sumexp) per lane, base-2 domain.
// --------------------------------------------------------------------------
__global__ void __launch_bounds__(128)
sinkhorn_pass(const _Float16* __restrict__ Rh, const _Float16* __restrict__ Ch,
              const float* __restrict__ r2, const float* __restrict__ c2,
              const float* __restrict__ pot_in, float* __restrict__ pot_out) {
  __shared__ float msh[4][16];
  __shared__ float ssh[4][16];

  const int lane = threadIdx.x & 31;
  const int wave = threadIdx.x >> 5;
  const int half = lane >> 4;
  const int l15  = lane & 15;
  const int row0 = blockIdx.x * 16;
  const int arow = row0 + l15;

  const v16h a0 = load_a(Rh, arow, 0,  half);
  const v16h a1 = load_a(Rh, arow, 32, half);

  float rr[8];
  #pragma unroll
  for (int e = 0; e < 8; ++e)
    rr[e] = r2[row0 + half * 8 + e] * (INV_EPS * L2E);

  float rm[8], rs[8];
  #pragma unroll
  for (int e = 0; e < 8; ++e) { rm[e] = -3.0e38f; rs[e] = 0.0f; }

  // 128 tiles per wave, two per iteration (ct and ct+4).
  for (int ct = wave; ct < NT; ct += 8) {
    const int na = ct * 16 + l15;
    const int nb = na + 64;                     // tile ct+4
    const v16h b0a = load_b(Ch, na, 0,  half);
    const v16h b1a = load_b(Ch, na, 32, half);
    const v16h b0b = load_b(Ch, nb, 0,  half);
    const v16h b1b = load_b(Ch, nb, 32, half);

    v8f accA = {}, accB = {};
    accA = __builtin_amdgcn_wmma_f32_16x16x32_f16(false, a0, false, b0a,
                                                  (short)0, accA, false, false);
    accA = __builtin_amdgcn_wmma_f32_16x16x32_f16(false, a1, false, b1a,
                                                  (short)0, accA, false, false);
    accB = __builtin_amdgcn_wmma_f32_16x16x32_f16(false, a0, false, b0b,
                                                  (short)0, accB, false, false);
    accB = __builtin_amdgcn_wmma_f32_16x16x32_f16(false, a1, false, b1b,
                                                  (short)0, accB, false, false);

    const float cca = fmaf(-c2[na], INV_EPS, pot_in[na]) * L2E;
    #pragma unroll
    for (int e = 0; e < 8; ++e) {
      const float v  = fmaf(accA[e], K2C, cca) - rr[e];
      const float nm = fmaxf(rm[e], v);
      rs[e] = rs[e] * fexp2(rm[e] - nm) + fexp2(v - nm);
      rm[e] = nm;
    }
    const float ccb = fmaf(-c2[nb], INV_EPS, pot_in[nb]) * L2E;
    #pragma unroll
    for (int e = 0; e < 8; ++e) {
      const float v  = fmaf(accB[e], K2C, ccb) - rr[e];
      const float nm = fmaxf(rm[e], v);
      rs[e] = rs[e] * fexp2(rm[e] - nm) + fexp2(v - nm);
      rm[e] = nm;
    }
  }

  // Reduce across the 16 lanes holding each row (xor masks stay in-half).
  #pragma unroll
  for (int mask = 1; mask <= 8; mask <<= 1) {
    #pragma unroll
    for (int e = 0; e < 8; ++e) {
      const float om = __shfl_xor(rm[e], mask, 32);
      const float os = __shfl_xor(rs[e], mask, 32);
      const float nm = fmaxf(rm[e], om);
      rs[e] = rs[e] * fexp2(rm[e] - nm) + os * fexp2(om - nm);
      rm[e] = nm;
    }
  }

  if (lane == 0 || lane == 16) {
    #pragma unroll
    for (int e = 0; e < 8; ++e) {
      msh[wave][half * 8 + e] = rm[e];
      ssh[wave][half * 8 + e] = rs[e];
    }
  }
  __syncthreads();

  if (threadIdx.x < 16) {
    const int t = threadIdx.x;
    float M = msh[0][t], S = ssh[0][t];
    #pragma unroll
    for (int w = 1; w < 4; ++w) {
      const float m2 = msh[w][t], s2 = ssh[w][t];
      const float nm = fmaxf(M, m2);
      S = S * fexp2(M - nm) + s2 * fexp2(m2 - nm);
      M = nm;
    }
    pot_out[row0 + t] = LOG_W - (M + flog2(fmaxf(S, 1e-37f))) * LN2;
  }
}

// --------------------------------------------------------------------------
// P[i,j] = exp(f_i + log_K[i,j] + g_j), recomputed with WMMA, streamed out.
// Same manual x2 tile unroll as the pass kernel.
// --------------------------------------------------------------------------
__global__ void __launch_bounds__(128)
write_P(const _Float16* __restrict__ Xh, const _Float16* __restrict__ Yh,
        const float* __restrict__ x2, const float* __restrict__ y2,
        const float* __restrict__ f, const float* __restrict__ g,
        float* __restrict__ P) {
  const int lane = threadIdx.x & 31;
  const int wave = threadIdx.x >> 5;
  const int half = lane >> 4;
  const int l15  = lane & 15;
  const int row0 = blockIdx.x * 16;
  const int arow = row0 + l15;

  const v16h a0 = load_a(Xh, arow, 0,  half);
  const v16h a1 = load_a(Xh, arow, 32, half);

  float rrp[8];
  #pragma unroll
  for (int e = 0; e < 8; ++e) {
    const int r = row0 + half * 8 + e;
    rrp[e] = (x2[r] * INV_EPS - f[r]) * L2E;
  }

  for (int ct = wave; ct < NT; ct += 8) {
    const int na = ct * 16 + l15;
    const int nb = na + 64;
    const v16h b0a = load_b(Yh, na, 0,  half);
    const v16h b1a = load_b(Yh, na, 32, half);
    const v16h b0b = load_b(Yh, nb, 0,  half);
    const v16h b1b = load_b(Yh, nb, 32, half);

    v8f accA = {}, accB = {};
    accA = __builtin_amdgcn_wmma_f32_16x16x32_f16(false, a0, false, b0a,
                                                  (short)0, accA, false, false);
    accA = __builtin_amdgcn_wmma_f32_16x16x32_f16(false, a1, false, b1a,
                                                  (short)0, accA, false, false);
    accB = __builtin_amdgcn_wmma_f32_16x16x32_f16(false, a0, false, b0b,
                                                  (short)0, accB, false, false);
    accB = __builtin_amdgcn_wmma_f32_16x16x32_f16(false, a1, false, b1b,
                                                  (short)0, accB, false, false);

    const float cca = fmaf(-y2[na], INV_EPS, g[na]) * L2E;
    #pragma unroll
    for (int e = 0; e < 8; ++e) {
      const size_t off = (size_t)(row0 + half * 8 + e) * N2 + na;
      P[off] = fexp2(fmaf(accA[e], K2C, cca) - rrp[e]);
    }
    const float ccb = fmaf(-y2[nb], INV_EPS, g[nb]) * L2E;
    #pragma unroll
    for (int e = 0; e < 8; ++e) {
      const size_t off = (size_t)(row0 + half * 8 + e) * N2 + nb;
      P[off] = fexp2(fmaf(accB[e], K2C, ccb) - rrp[e]);
    }
  }
}

// --------------------------------------------------------------------------
// aligned[j,:] = (sum_i P[i,j] * X[i,:]) / max(sum_i P[i,j], 1e-30)
// Block owns 16 columns j; thread (jl, dc) accumulates 4 of the 64 dims.
// --------------------------------------------------------------------------
__global__ void __launch_bounds__(256)
col_aligned(const float* __restrict__ P, const float* __restrict__ X,
            float* __restrict__ out) {
  const int jl = threadIdx.x >> 4;        // 0..15 column within block
  const int dc = threadIdx.x & 15;        // 0..15 chunk of 4 dims
  const int j  = blockIdx.x * 16 + jl;
  const float4* __restrict__ X4 = (const float4*)X;

  float ax = 0.f, ay = 0.f, az = 0.f, aw = 0.f, cm = 0.f;
  for (int i = 0; i < N1; i += 4) {
    #pragma unroll
    for (int u = 0; u < 4; ++u) {
      const float  p  = P[(size_t)(i + u) * N2 + j];
      const float4 xv = X4[(i + u) * (D / 4) + dc];
      ax = fmaf(p, xv.x, ax);
      ay = fmaf(p, xv.y, ay);
      az = fmaf(p, xv.z, az);
      aw = fmaf(p, xv.w, aw);
      cm += p;
    }
  }
  __shared__ float cms[16];
  if (dc == 0) cms[jl] = fmaxf(cm, 1e-30f);
  __syncthreads();
  const float inv = 1.0f / cms[jl];
  float* o = out + (size_t)j * D + dc * 4;
  o[0] = ax * inv; o[1] = ay * inv; o[2] = az * inv; o[3] = aw * inv;
}

// --------------------------------------------------------------------------
extern "C" void kernel_launch(void* const* d_in, const int* in_sizes, int n_in,
                              void* d_out, int out_size, void* d_ws, size_t ws_size,
                              hipStream_t stream) {
  (void)in_sizes; (void)n_in; (void)out_size; (void)ws_size;
  const float* X = (const float*)d_in[0];
  const float* Y = (const float*)d_in[1];

  char* ws = (char*)d_ws;
  _Float16* Xh = (_Float16*)ws;                         // 1 MiB
  _Float16* Yh = (_Float16*)(ws + (1 << 20));           // 1 MiB
  float* x2 = (float*)(ws + (2 << 20));                 // 32 KiB
  float* y2 = x2 + N1;
  float* f  = y2 + N2;
  float* g  = f + N1;

  prep_kernel<<<(N1 + N2) / 256, 256, 0, stream>>>(X, Y, Xh, Yh, x2, y2, g);

  for (int it = 0; it < SINK_ITERS; ++it) {
    // f = log_a - LSE_j(log_K + g)   (rows = X)
    sinkhorn_pass<<<N1 / 16, 128, 0, stream>>>(Xh, Yh, x2, y2, g, f);
    // g = log_b - LSE_i(log_K + f)   (rows = Y, log_K transposed by symmetry)
    sinkhorn_pass<<<N2 / 16, 128, 0, stream>>>(Yh, Xh, y2, x2, f, g);
  }

  float* P       = (float*)d_out;
  float* aligned = P + (size_t)N1 * N2;
  write_P<<<N1 / 16, 128, 0, stream>>>(Xh, Yh, x2, y2, f, g, P);
  col_aligned<<<N2 / 16, 256, 0, stream>>>(P, X, aligned);
}
